// SNNModelTF_11493332484486
// MI455X (gfx1250) — matrix operand order repaired and checked by reading, not compiled, not run on hardware
//
#include <hip/hip_runtime.h>

// ---------------------------------------------------------------------------
// SNN forward for MI455X (gfx1250, wave32, WMMA).
//   Layer0: (B*T x IN) @ (IN x H0) via v_wmma_f32_16x16x32_f16, fused with
//           the per-timestep LIF scan. B tile (weights) staged in LDS once
//           per block (invariant over t), each wave computes a 16x64 tile
//           -> 4 WMMAs per A load, ~400x reuse of the staged tile.
//   Layers 1..3: counts @ W via WMMA, then constant-current LIF.
//   Output: c3 @ Wout + bout (WMMA, no LIF).
// ---------------------------------------------------------------------------

#define NB   512      // batch
#define NT   50       // timesteps
#define NIN  1024
#define NH0  2048
#define NH1  1024
#define NOUT 1024
#define TAU   0.25f
#define VTH   1.0f

typedef __attribute__((ext_vector_type(16))) _Float16 v16h;
typedef __attribute__((ext_vector_type(8)))  _Float16 v8h;
typedef __attribute__((ext_vector_type(8)))  float    v8f;

// ---------------------------------------------------------------------------
// f32 -> f16 conversion (x spike train). n must be a multiple of 4.
// ---------------------------------------------------------------------------
__global__ void snn_f32_to_f16(const float* __restrict__ src,
                               _Float16* __restrict__ dst, int n) {
    int i = (blockIdx.x * blockDim.x + threadIdx.x) * 4;
    if (i + 3 < n) {
        dst[i + 0] = (_Float16)src[i + 0];
        dst[i + 1] = (_Float16)src[i + 1];
        dst[i + 2] = (_Float16)src[i + 2];
        dst[i + 3] = (_Float16)src[i + 3];
    }
}

// ---------------------------------------------------------------------------
// Pack a row-major (K x N) f32 weight matrix into the per-lane B-operand
// layout of V_WMMA_F32_16X16X32_F16:
//   tile index (n_tile, k_step); lane 0..31; 16 contiguous halves per lane.
//   lane<16  : column n_tile*16+lane,      K = k_step*32 + 0..7, 16..23
//   lane>=16 : column n_tile*16+lane-16,   K = k_step*32 + 8..15, 24..31
// ---------------------------------------------------------------------------
__global__ void snn_pack_b(const float* __restrict__ W,
                           _Float16* __restrict__ Wp, int K, int N) {
    int tid   = blockIdx.x * blockDim.x + threadIdx.x;
    int ksteps = K >> 5;
    int total  = (N >> 4) * ksteps * 32;
    if (tid >= total) return;
    int lane   = tid & 31;
    int k_step = (tid >> 5) % ksteps;
    int n_tile = tid / (32 * ksteps);
    int n  = n_tile * 16 + (lane & 15);
    int kb = k_step * 32 + ((lane >> 4) << 3);
    _Float16* dst = Wp + (size_t)tid * 16;
#pragma unroll
    for (int kk = 0; kk < 16; ++kk)
        dst[kk] = (_Float16)W[(size_t)(kb + kk + (kk >= 8 ? 8 : 0)) * N + n];
}

// Build the A operand (16x32 f16) for one K-step from a row pointer.
// Lane<16 holds M=lane,  K = 0..7 and 16..23; lane>=16 holds K=8..15,24..31.
__device__ __forceinline__ v16h load_a_f16(const _Float16* row, int laneHi) {
    v8h a0 = *(const v8h*)(row + laneHi * 8);
    v8h a1 = *(const v8h*)(row + 16 + laneHi * 8);
    v16h a;
#pragma unroll
    for (int i = 0; i < 8; ++i) { a[i] = a0[i]; a[i + 8] = a1[i]; }
    return a;
}

// ---------------------------------------------------------------------------
// Fused layer 0, LDS-staged B, 16x64 tile per wave.
// grid.x = NH0/64 column blocks, grid.y = NB/128; block = 256 (8 waves).
// Dynamic LDS: 4 * NIN * 16 halves = 128 KB (B tile for 64 columns).
// ---------------------------------------------------------------------------
__global__ void __launch_bounds__(256)
snn_lif0_wmma(const _Float16* __restrict__ x_h,     // (NB*NT, NIN) f16
              const _Float16* __restrict__ Bp,      // packed W0
              const float*    __restrict__ bias,    // b0 (NH0)
              float*          __restrict__ s0_out,  // (NB, NT, NH0)
              _Float16*       __restrict__ c0_out)  // (NB, NH0) counts
{
    extern __shared__ _Float16 Blds[];              // 4 * NIN * 16 halves

    const int wave   = threadIdx.x >> 5;
    const int lane   = threadIdx.x & 31;
    const int laneN  = lane & 15;
    const int laneHi = lane >> 4;

    const int o_base = blockIdx.x * 64;
    const int b_base = blockIdx.y * 128 + wave * 16;
    const int b_row  = b_base + laneN;             // A-matrix row for this lane
    const int b_out  = b_base + laneHi * 8;        // C-matrix row base (VGPR 0)

    // Cooperative stage of 4 consecutive packed o-tiles (contiguous in Bp).
    {
        const _Float16* gB = Bp + (size_t)blockIdx.x * 4 * NIN * 16;
        const int nvec = 4 * NIN * 16 / 8;          // v8h chunks
        for (int i = threadIdx.x; i < nvec; i += blockDim.x)
            *(v8h*)(Blds + (size_t)i * 8) = *(const v8h*)(gB + (size_t)i * 8);
    }
    __syncthreads();

    float bias_v[4];
#pragma unroll
    for (int j = 0; j < 4; ++j) bias_v[j] = bias[o_base + j * 16 + laneN];

    v8f vmem[4] = {}, cnt[4] = {};

    for (int t = 0; t < NT; ++t) {
        v8f acc[4];
#pragma unroll
        for (int j = 0; j < 4; ++j)
#pragma unroll
            for (int r = 0; r < 8; ++r) acc[j][r] = bias_v[j];

        const _Float16* arow = x_h + ((size_t)b_row * NT + t) * NIN;
#pragma unroll 2
        for (int ks = 0; ks < NIN / 32; ++ks) {
            __builtin_prefetch(arow + ks * 32 + 256, 0, 1);
            v16h a = load_a_f16(arow + ks * 32, laneHi);
#pragma unroll
            for (int j = 0; j < 4; ++j) {
                v16h b = *(const v16h*)(Blds + (size_t)j * NIN * 16 +
                                        ((size_t)ks * 32 + lane) * 16);
                acc[j] = __builtin_amdgcn_wmma_f32_16x16x32_f16(
                    false, a, false, b, (short)0, acc[j], false, false);
            }
        }
        // LIF update in accumulator layout: M = r + 8*laneHi, N = laneN.
#pragma unroll
        for (int j = 0; j < 4; ++j)
#pragma unroll
            for (int r = 0; r < 8; ++r) {
                vmem[j][r] = TAU * vmem[j][r] + (1.0f - TAU) * acc[j][r];
                float s = (vmem[j][r] > VTH) ? 1.0f : 0.0f;
                vmem[j][r] -= s * VTH;
                cnt[j][r]  += s;
                s0_out[((size_t)(b_out + r) * NT + t) * NH0 +
                       o_base + j * 16 + laneN] = s;
            }
    }
#pragma unroll
    for (int j = 0; j < 4; ++j)
#pragma unroll
        for (int r = 0; r < 8; ++r)
            c0_out[(size_t)(b_out + r) * NH0 + o_base + j * 16 + laneN] =
                (_Float16)cnt[j][r];
}

// ---------------------------------------------------------------------------
// Generic WMMA GEMM over counts: (NB x K) @ (K x N) + bias, then either
//   do_lif=1: constant-current LIF for NT steps -> spikes (optional) + counts
//   do_lif=0: store accumulator to out_f32.
// grid.x = N/16, grid.y = NB/128; block = 256.
// ---------------------------------------------------------------------------
__global__ void __launch_bounds__(256)
snn_gemm_lif_wmma(const _Float16* __restrict__ A_h,     // (NB, K) f16 counts
                  const _Float16* __restrict__ Bp,      // packed weights
                  const float*    __restrict__ bias,
                  float*          __restrict__ s_out,   // may be nullptr
                  _Float16*       __restrict__ cnt_out, // counts (if do_lif)
                  float*          __restrict__ out_f32, // (if !do_lif)
                  int K, int N, int do_lif)
{
    const int wave   = threadIdx.x >> 5;
    const int lane   = threadIdx.x & 31;
    const int laneN  = lane & 15;
    const int laneHi = lane >> 4;

    const int o_base = blockIdx.x * 16;
    const int b_base = blockIdx.y * 128 + wave * 16;
    const int b_row  = b_base + laneN;
    const int b_out  = b_base + laneHi * 8;

    const _Float16* bp_tile = Bp + (size_t)blockIdx.x * K * 16;
    const float bias_v = bias[o_base + laneN];

    v8f acc;
#pragma unroll
    for (int r = 0; r < 8; ++r) acc[r] = bias_v;

    const _Float16* arow = A_h + (size_t)b_row * K;
#pragma unroll 4
    for (int ks = 0; ks < (K >> 5); ++ks) {
        v16h a = load_a_f16(arow + ks * 32, laneHi);
        const _Float16* bptr = bp_tile + ((size_t)ks * 32 + lane) * 16;
        __builtin_prefetch(bptr + 32 * 16, 0, 1);
        v16h b = *(const v16h*)bptr;
        acc = __builtin_amdgcn_wmma_f32_16x16x32_f16(
            false, a, false, b, (short)0, acc, false, false);
    }

    if (!do_lif) {
#pragma unroll
        for (int r = 0; r < 8; ++r)
            out_f32[(size_t)(b_out + r) * N + o_base + laneN] = acc[r];
        return;
    }

#pragma unroll
    for (int r = 0; r < 8; ++r) {
        const float c = acc[r];
        float v = 0.0f, count = 0.0f;
        float* srow = s_out ? s_out + (size_t)(b_out + r) * NT * N
                              + o_base + laneN : nullptr;
        for (int t = 0; t < NT; ++t) {
            v = TAU * v + (1.0f - TAU) * c;
            float s = (v > VTH) ? 1.0f : 0.0f;
            v -= s * VTH;
            count += s;
            if (srow) srow[(size_t)t * N] = s;
        }
        cnt_out[(size_t)(b_out + r) * N + o_base + laneN] = (_Float16)count;
    }
}

// ---------------------------------------------------------------------------
// Host-side launch.
// d_in: x, W0, b0, W1, b1, W2, b2, Wout, bout
// d_out: out(512*1024) | s0(512*50*2048) | s1(512*50*1024) | s2(512*50*1024)
// ---------------------------------------------------------------------------
extern "C" void kernel_launch(void* const* d_in, const int* in_sizes, int n_in,
                              void* d_out, int out_size, void* d_ws, size_t ws_size,
                              hipStream_t stream) {
    const float* x    = (const float*)d_in[0];
    const float* W0   = (const float*)d_in[1];
    const float* b0   = (const float*)d_in[2];
    const float* W1   = (const float*)d_in[3];
    const float* b1   = (const float*)d_in[4];
    const float* W2   = (const float*)d_in[5];
    const float* b2   = (const float*)d_in[6];
    const float* Wout = (const float*)d_in[7];
    const float* bout = (const float*)d_in[8];

    float* out = (float*)d_out;
    float* s0  = out + (size_t)NB * NOUT;                 // 512*1024
    float* s1  = s0  + (size_t)NB * NT * NH0;             // +512*50*2048
    float* s2  = s1  + (size_t)NB * NT * NH1;             // +512*50*1024

    // Workspace carve-up (all 256-byte aligned sizes).
    char* ws = (char*)d_ws;
    _Float16* x_h   = (_Float16*)ws;                       ws += (size_t)NB * NT * NIN * 2;
    _Float16* W0p   = (_Float16*)ws;                       ws += (size_t)NIN * NH0 * 2;
    _Float16* W1p   = (_Float16*)ws;                       ws += (size_t)NH0 * NH1 * 2;
    _Float16* W2p   = (_Float16*)ws;                       ws += (size_t)NH1 * NOUT * 2;
    _Float16* Woutp = (_Float16*)ws;                       ws += (size_t)NOUT * NOUT * 2;
    _Float16* c0_h  = (_Float16*)ws;                       ws += (size_t)NB * NH0 * 2;
    _Float16* c1_h  = (_Float16*)ws;                       ws += (size_t)NB * NH1 * 2;
    _Float16* c2_h  = (_Float16*)ws;                       ws += (size_t)NB * NOUT * 2;
    _Float16* c3_h  = (_Float16*)ws;

    // 1) x -> f16
    {
        int n = NB * NT * NIN;
        snn_f32_to_f16<<<n / 1024, 256, 0, stream>>>(x, x_h, n);
    }
    // 2) pack weights into WMMA B-operand layout
    {
        int t0 = (NH0 / 16) * (NIN / 32) * 32;   // 131072
        snn_pack_b<<<(t0 + 255) / 256, 256, 0, stream>>>(W0, W0p, NIN, NH0);
        int t1 = (NH1 / 16) * (NH0 / 32) * 32;   // 131072
        snn_pack_b<<<(t1 + 255) / 256, 256, 0, stream>>>(W1, W1p, NH0, NH1);
        int t2 = (NOUT / 16) * (NH1 / 32) * 32;  // 65536
        snn_pack_b<<<(t2 + 255) / 256, 256, 0, stream>>>(W2, W2p, NH1, NOUT);
        snn_pack_b<<<(t2 + 255) / 256, 256, 0, stream>>>(Wout, Woutp, NOUT, NOUT);
    }
    // 3) fused layer 0 (the 107-GFLOP GEMM + LIF scan), B staged in LDS
    snn_lif0_wmma<<<dim3(NH0 / 64, NB / 128), 256,
                    4 * NIN * 16 * sizeof(_Float16), stream>>>(
        x_h, W0p, b0, s0, c0_h);
    // 4) layer 1: c0 @ W1 + b1 -> LIF -> s1, c1
    snn_gemm_lif_wmma<<<dim3(NH1 / 16, NB / 128), 256, 0, stream>>>(
        c0_h, W1p, b1, s1, c1_h, nullptr, NH0, NH1, 1);
    // 5) layer 2: c1 @ W2 + b2 -> LIF -> s2, c2
    snn_gemm_lif_wmma<<<dim3(NOUT / 16, NB / 128), 256, 0, stream>>>(
        c1_h, W2p, b2, s2, c2_h, nullptr, NH1, NOUT, 1);
    // 6) layer 3 (re-applies W2/b2, spikes discarded) -> c3
    snn_gemm_lif_wmma<<<dim3(NOUT / 16, NB / 128), 256, 0, stream>>>(
        c2_h, W2p, b2, nullptr, c3_h, nullptr, NH1, NOUT, 1);
    // 7) out = c3 @ Wout + bout
    snn_gemm_lif_wmma<<<dim3(NOUT / 16, NB / 128), 256, 0, stream>>>(
        c3_h, Woutp, bout, nullptr, nullptr, out, NOUT, NOUT, 0);
}
